// GapJunctionNetwork_57664230916233
// MI455X (gfx1250) — matrix operand order repaired and checked by reading, not compiled, not run on hardware
//
#include <hip/hip_runtime.h>

#define Nn 1024
#define Dd 64

typedef __attribute__((ext_vector_type(16))) _Float16 v16h;
typedef __attribute__((ext_vector_type(8)))  float    v8f;
typedef __attribute__((ext_vector_type(2)))  _Float16 h2v;
typedef __attribute__((ext_vector_type(4)))  uint32_t u32x4;

union HU { v16h h; h2v p[8]; uint32_t u[8]; };
union QU { u32x4 q; h2v p[4]; uint32_t u[4]; };
union QF { u32x4 q; float f[4]; };

static __device__ inline h2v relu2(h2v x) {
#if __has_builtin(__builtin_elementwise_max)
    const h2v z = {(_Float16)0.0f, (_Float16)0.0f};
    return __builtin_elementwise_max(x, z);
#else
    h2v r;
    r[0] = x[0] > (_Float16)0.0f ? x[0] : (_Float16)0.0f;
    r[1] = x[1] > (_Float16)0.0f ? x[1] : (_Float16)0.0f;
    return r;
#endif
}

static __device__ inline float reluf(float x) {
#if __has_builtin(__builtin_amdgcn_fmed3f)
    return __builtin_amdgcn_fmed3f(x, 0.0f, 3.0e38f);   // one v_med3_f32
#else
    return fmaxf(x, 0.0f);
#endif
}

static __device__ inline float fast_rcp(float x) {
#if __has_builtin(__builtin_amdgcn_rcpf)
    return __builtin_amdgcn_rcpf(x);
#else
    return 1.0f / x;
#endif
}

// ---------------------------------------------------------------------------
// K1: hi = cs@w1a^T (f16), hjp = cs@w1b^T + b1 (f16), trT = (cs@sig_w^T+b)^T (f16)
// ---------------------------------------------------------------------------
__global__ __launch_bounds__(64)
void k_pre(const float* __restrict__ cs, const float* __restrict__ gw1,
           const float* __restrict__ gb1, const float* __restrict__ sw,
           const float* __restrict__ sb,
           _Float16* __restrict__ hi, _Float16* __restrict__ hjp,
           _Float16* __restrict__ trT)
{
    __shared__ float row[Dd];
    const int i = blockIdx.x, t = threadIdx.x;
    row[t] = cs[i * Dd + t];
    __syncthreads();
    float a = 0.f, b = 0.f, c = 0.f;
    for (int k = 0; k < Dd; ++k) {
        const float x = row[k];
        a += x * gw1[t * 128 + k];
        b += x * gw1[t * 128 + 64 + k];
        c += x * sw[t * Dd + k];
    }
    hi[i * Dd + t]  = (_Float16)a;
    hjp[i * Dd + t] = (_Float16)(b + gb1[t]);
    trT[t * Nn + i] = (_Float16)(c + sb[t]);
}

// ---------------------------------------------------------------------------
// K2: sym = 0.5*(A + A^T), zero diagonal
// ---------------------------------------------------------------------------
__global__ __launch_bounds__(256)
void k_sym(const float* __restrict__ A, float* __restrict__ sym)
{
    __shared__ float at[64][65];
    const int bi = blockIdx.x * 64, bj = blockIdx.y * 64;
    const int tid = threadIdx.x;
    for (int e = tid; e < 4096; e += 256) {
        const int r = e >> 6, c = e & 63;
        at[r][c] = A[(bj + r) * Nn + bi + c];
    }
    __syncthreads();
    for (int e = tid; e < 4096; e += 256) {
        const int r = e >> 6, c = e & 63;
        const int i = bi + r, j = bj + c;
        sym[i * Nn + j] = (i == j) ? 0.f : 0.5f * (A[i * Nn + j] + at[c][r]);
    }
}

// ---------------------------------------------------------------------------
// K3: per-row softmax stats
// ---------------------------------------------------------------------------
__global__ __launch_bounds__(256)
void k_stats(const float* __restrict__ sym, float* __restrict__ rmax,
             float* __restrict__ rsum)
{
    __shared__ float red[256];
    const int i = blockIdx.x, t = threadIdx.x;
    float m = -3.4e38f;
    for (int j = t; j < Nn; j += 256) m = fmaxf(m, sym[i * Nn + j]);
    red[t] = m; __syncthreads();
    for (int s = 128; s > 0; s >>= 1) {
        if (t < s) red[t] = fmaxf(red[t], red[t + s]);
        __syncthreads();
    }
    const float mm = red[0];
    __syncthreads();
    float sa = 0.f;
    for (int j = t; j < Nn; j += 256) sa += __expf(sym[i * Nn + j] - mm);
    red[t] = sa; __syncthreads();
    for (int s = 128; s > 0; s >>= 1) {
        if (t < s) red[t] += red[t + s];
        __syncthreads();
    }
    if (t == 0) { rmax[i] = mm; rsum[i] = red[0]; }
}

// ---------------------------------------------------------------------------
// K4: fused pairwise gate MLP + weighted transfer GEMM (f16 WMMA, f32 acc)
//   - gate GEMM transposed (h2^T = w2 @ h1^T), lane-local w3 dot, 1 shfl/j
//   - j loop unrolled x4 (independent WMMA chains hide RAW hazard slots)
//   - hi fragments hoisted to registers; hj chunk staged via
//     global_load_async_to_lds_b128 (+ s_wait_asynccnt), per CDNA5 ISA
// ---------------------------------------------------------------------------
__global__ __launch_bounds__(256)
void k_main(const _Float16* __restrict__ hi, const _Float16* __restrict__ hjp,
            const _Float16* __restrict__ trT, const float* __restrict__ sym,
            const float* __restrict__ rmax, const float* __restrict__ rsum,
            const float* __restrict__ gw2, const float* __restrict__ gb2,
            const float* __restrict__ gw3, const float* __restrict__ gb3,
            float* __restrict__ partial)
{
    __shared__ __align__(16) char smem[47488];
    _Float16* hiS   = (_Float16*)smem;             // [16][72] f16
    _Float16* w2S   = (_Float16*)(smem + 2304);    // [32][64] f16
    float*    rmaxS = (float*)(smem + 6400);       // [16]
    float*    rsumS = (float*)(smem + 6464);       // [16]
    _Float16* hjS   = (_Float16*)(smem + 6528);    // [8][32][64] f16 (per wave)
    _Float16* wgtS  = (_Float16*)(smem + 39296);   // [8][32][16] f16 (per wave)
    float*    partS = (float*)smem;                // phase 2: [8][16][64] f32

    const int tid  = threadIdx.x;
    const int wv   = tid >> 5;
    const int lane = tid & 31;
    const int hl   = lane >> 4;
    const int n    = lane & 15;
    const int ibase = blockIdx.x * 16;
    const int jbase = blockIdx.y * 256 + wv * 32;

    _Float16* hjW  = hjS  + wv * 2048;
    _Float16* wgtW = wgtS + wv * 512;

    // ---- async stage of this wave's hj chunk (32 rows x 64 f16 = 4 KB) ---
    {
        const uint64_t gbase = (uint64_t)(uintptr_t)(hjp + (size_t)jbase * Dd);
        const uint32_t lbase = (uint32_t)(uintptr_t)hjW;
        for (int e = lane; e < 256; e += 32) {
            const uint32_t loff = lbase + (uint32_t)e * 16u;
            const uint32_t goff = (uint32_t)e * 16u;
            asm volatile("global_load_async_to_lds_b128 %0, %1, %2"
                         :: "v"(loff), "v"(goff), "s"(gbase) : "memory");
        }
    }

    // ---- synchronous staging of shared tiles -----------------------------
    {
        const uint32_t* src = (const uint32_t*)(hi + ibase * Dd);
        uint32_t*       dst = (uint32_t*)hiS;
        for (int e = tid; e < 512; e += 256) {
            const int r = e >> 5, c = e & 31;
            dst[r * 36 + c] = src[r * 32 + c];
        }
    }
    for (int e = tid; e < 1024; e += 256) {
        h2v w; w[0] = (_Float16)gw2[2 * e]; w[1] = (_Float16)gw2[2 * e + 1];
        *(h2v*)&w2S[2 * e] = w;
    }
    if (tid < 16) { rmaxS[tid] = rmax[ibase + tid]; rsumS[tid] = rsum[ibase + tid]; }

    asm volatile("s_wait_asynccnt 0" ::: "memory");
    __syncthreads();

    // ---- register A-fragments of w2: 2 m-tiles x 2 k-steps ---------------
    HU Aw2[2][2];
    #pragma unroll
    for (int mt = 0; mt < 2; ++mt)
        #pragma unroll
        for (int s = 0; s < 2; ++s) {
            const _Float16* rowp = w2S + (n + 16 * mt) * 64 + 16 * hl + 32 * s;
            QU a0, a1;
            a0.q = *(const u32x4*)rowp;
            a1.q = *(const u32x4*)(rowp + 8);
            #pragma unroll
            for (int v = 0; v < 4; ++v) {
                Aw2[mt][s].p[v]     = a0.p[v];
                Aw2[mt][s].p[4 + v] = a1.p[v];
            }
        }

    // hi fragment halves: loop-invariant, keep in registers
    QU hiR[2][2];
    #pragma unroll
    for (int s = 0; s < 2; ++s) {
        const _Float16* hp = hiS + n * 72 + 16 * hl + 32 * s;
        hiR[s][0].q = *(const u32x4*)hp;
        hiR[s][1].q = *(const u32x4*)(hp + 8);
    }

    float w3r0[8], w3r1[8];
    v8f cb0, cb1;
    #pragma unroll
    for (int r = 0; r < 8; ++r) {
        w3r0[r] = gw3[r + 8 * hl];
        w3r1[r] = gw3[r + 8 * hl + 16];
        cb0[r]  = gb2[r + 8 * hl];
        cb1[r]  = gb2[r + 8 * hl + 16];
    }
    const float b3v  = gb3[0];
    const float rm   = rmaxS[n];
    const float invs = fast_rcp(rsumS[n]);
    const int   ig   = ibase + n;
    const float* symRowC = sym + ig * Nn + jbase;

    v8f acc[4];
    #pragma unroll
    for (int nt = 0; nt < 4; ++nt)
        #pragma unroll
        for (int r = 0; r < 8; ++r) acc[nt][r] = 0.f;

    // ---- main loop: quads of 4 independent j -----------------------------
    for (int q = 0; q < 8; ++q) {
        QF s4;
        s4.q = *(const u32x4*)(symRowC + 4 * q);   // sym row slice for 4 j's
        #pragma unroll
        for (int u = 0; u < 4; ++u) {
            const int jj = 4 * q + u;
            const int jg = jbase + jj;

            // B = h1^T fragment (K = feature, N = pair i), packed f16 math
            HU Bh[2];
            #pragma unroll
            for (int s = 0; s < 2; ++s) {
                const _Float16* jp = hjW + jj * 64 + 16 * hl + 32 * s;
                QU j0, j1;
                j0.q = *(const u32x4*)jp;
                j1.q = *(const u32x4*)(jp + 8);
                #pragma unroll
                for (int v = 0; v < 4; ++v) {
                    Bh[s].p[v]     = relu2(hiR[s][0].p[v] + j0.p[v]);
                    Bh[s].p[4 + v] = relu2(hiR[s][1].p[v] + j1.p[v]);
                }
            }

            // h2^T = w2 @ h1^T + b2
            v8f D0 = cb0, D1 = cb1;
            D0 = __builtin_amdgcn_wmma_f32_16x16x32_f16(false, Aw2[0][0].h, false, Bh[0].h, (short)0, D0, false, false);
            D0 = __builtin_amdgcn_wmma_f32_16x16x32_f16(false, Aw2[0][1].h, false, Bh[1].h, (short)0, D0, false, false);
            D1 = __builtin_amdgcn_wmma_f32_16x16x32_f16(false, Aw2[1][0].h, false, Bh[0].h, (short)0, D1, false, false);
            D1 = __builtin_amdgcn_wmma_f32_16x16x32_f16(false, Aw2[1][1].h, false, Bh[1].h, (short)0, D1, false, false);

            // gate = sigmoid(relu(h2) . w3 + b3): lane-local, one shfl
            float t = 0.f;
            #pragma unroll
            for (int r = 0; r < 8; ++r) {
                t += reluf(D0[r]) * w3r0[r];
                t += reluf(D1[r]) * w3r1[r];
            }
            t += __shfl_xor(t, 16, 32);

            const float gate = fast_rcp(1.f + __expf(-(t + b3v)));
            float w = 0.f;
            if (ig != jg)
                w = gate * __expf(s4.f[u] - rm) * invs;
            if (hl == 0) wgtW[jj * 16 + n] = (_Float16)w;
        }
    }

#if __has_builtin(__builtin_amdgcn_wave_barrier)
    __builtin_amdgcn_wave_barrier();
#endif

    // ---- out += weight(16x32) @ transferred(32x64) -----------------------
    v16h Aw;
    #pragma unroll
    for (int h = 0; h < 16; ++h) {
        const int k = h + 16 * hl;
        Aw[h] = wgtW[k * 16 + n];
    }
    #pragma unroll
    for (int nt = 0; nt < 4; ++nt) {
        HU Bt;
        const int d = nt * 16 + n;
        const _Float16* trp = trT + d * Nn + jbase + 16 * hl;
        QU b0, b1;
        b0.q = *(const u32x4*)trp;
        b1.q = *(const u32x4*)(trp + 8);
        #pragma unroll
        for (int v = 0; v < 4; ++v) { Bt.p[v] = b0.p[v]; Bt.p[4 + v] = b1.p[v]; }
        acc[nt] = __builtin_amdgcn_wmma_f32_16x16x32_f16(false, Aw, false, Bt.h, (short)0, acc[nt], false, false);
    }

    // ---- deterministic cross-wave reduction ------------------------------
    __syncthreads();
    #pragma unroll
    for (int nt = 0; nt < 4; ++nt)
        #pragma unroll
        for (int r = 0; r < 8; ++r)
            partS[wv * 1024 + (r + 8 * hl) * 64 + nt * 16 + n] = acc[nt][r];
    __syncthreads();
    for (int e = tid; e < 1024; e += 256) {
        float s = 0.f;
        #pragma unroll
        for (int w = 0; w < 8; ++w) s += partS[w * 1024 + e];
        partial[blockIdx.y * (Nn * Dd) + ibase * Dd + e] = s;
    }
}

// ---------------------------------------------------------------------------
// K5: out = cs + sum of 4 j-split partials
// ---------------------------------------------------------------------------
__global__ __launch_bounds__(256)
void k_out(const float* __restrict__ cs, const float* __restrict__ part,
           float* __restrict__ out)
{
    const int e = blockIdx.x * 256 + threadIdx.x;
    float s = cs[e];
    #pragma unroll
    for (int p = 0; p < 4; ++p) s += part[p * (Nn * Dd) + e];
    out[e] = s;
}

// ---------------------------------------------------------------------------
extern "C" void kernel_launch(void* const* d_in, const int* in_sizes, int n_in,
                              void* d_out, int out_size, void* d_ws, size_t ws_size,
                              hipStream_t stream)
{
    const float* cs  = (const float*)d_in[0];
    const float* adj = (const float*)d_in[1];
    const float* gw1 = (const float*)d_in[2];
    const float* gb1 = (const float*)d_in[3];
    const float* gw2 = (const float*)d_in[4];
    const float* gb2 = (const float*)d_in[5];
    const float* gw3 = (const float*)d_in[6];
    const float* gb3 = (const float*)d_in[7];
    const float* sw  = (const float*)d_in[8];
    const float* sb  = (const float*)d_in[9];
    float* out = (float*)d_out;

    char* ws = (char*)d_ws;
    _Float16* hi   = (_Float16*)ws;                // 1024*64 f16
    _Float16* hjp  = (_Float16*)(ws + 131072);     // 1024*64 f16
    _Float16* trT  = (_Float16*)(ws + 262144);     // 64*1024 f16
    float*    sym  = (float*)(ws + 393216);        // 1024*1024 f32
    float*    rmx  = (float*)(ws + 4587520);       // 1024 f32
    float*    rsm  = (float*)(ws + 4591616);       // 1024 f32
    float*    part = (float*)(ws + 4595712);       // 4*1024*64 f32

    k_pre  <<<Nn, 64, 0, stream>>>(cs, gw1, gb1, sw, sb, hi, hjp, trT);
    k_sym  <<<dim3(16, 16), 256, 0, stream>>>(adj, sym);
    k_stats<<<Nn, 256, 0, stream>>>(sym, rmx, rsm);
    k_main <<<dim3(64, 4), 256, 0, stream>>>(hi, hjp, trT, sym, rmx, rsm,
                                             gw2, gb2, gw3, gb3, part);
    k_out  <<<256, 256, 0, stream>>>(cs, part, out);
}
